// ZCLoRALinear_66486093742224
// MI455X (gfx1250) — matrix operand order repaired and checked by reading, not compiled, not run on hardware
//
#include <hip/hip_runtime.h>

// ---------------- CDNA5 types ----------------
typedef __attribute__((ext_vector_type(16))) __bf16 v16bf;
typedef __attribute__((ext_vector_type(8)))  __bf16 v8bf;
typedef __attribute__((ext_vector_type(4)))  __bf16 v4bf;
typedef __attribute__((ext_vector_type(8)))  float  v8f;

union Frag {
    v16bf v;
    v8bf  h[2];
};

#define BM   128
#define BN   128
#define BK   32
#define LDK  40   // padded LDS K-stride (bf16 elems): 80B rows, 16B-aligned, 20-bank step

// --------------------------------------------------------------------------
// Kernel 1: rank-4 projection t[s][r] = sum_i x[s,i]*B[r,i], and rs[s]=sum_i x[s,i]
// --------------------------------------------------------------------------
__global__ __launch_bounds__(256) void zclora_rank4(
    const float* __restrict__ x, const float* __restrict__ Bm,
    float* __restrict__ tb, float* __restrict__ rsb, int K)
{
    const int s   = blockIdx.x;
    const int tid = threadIdx.x;
    const float* xr = x + (long)s * K;

    float p0 = 0.f, p1 = 0.f, p2 = 0.f, p3 = 0.f, ps = 0.f;
    for (int i = tid * 4; i < K; i += 256 * 4) {
        float4 xv = *(const float4*)(xr + i);
        float4 b0 = *(const float4*)(Bm + 0 * K + i);
        float4 b1 = *(const float4*)(Bm + 1 * K + i);
        float4 b2 = *(const float4*)(Bm + 2 * K + i);
        float4 b3 = *(const float4*)(Bm + 3 * K + i);
        p0 += xv.x * b0.x + xv.y * b0.y + xv.z * b0.z + xv.w * b0.w;
        p1 += xv.x * b1.x + xv.y * b1.y + xv.z * b1.z + xv.w * b1.w;
        p2 += xv.x * b2.x + xv.y * b2.y + xv.z * b2.z + xv.w * b2.w;
        p3 += xv.x * b3.x + xv.y * b3.y + xv.z * b3.z + xv.w * b3.w;
        ps += xv.x + xv.y + xv.z + xv.w;
    }
    // wave32 reduction
    #pragma unroll
    for (int off = 16; off > 0; off >>= 1) {
        p0 += __shfl_xor(p0, off, 32);
        p1 += __shfl_xor(p1, off, 32);
        p2 += __shfl_xor(p2, off, 32);
        p3 += __shfl_xor(p3, off, 32);
        ps += __shfl_xor(ps, off, 32);
    }
    __shared__ float red[8][5];
    const int lane = tid & 31, wave = tid >> 5;
    if (lane == 0) {
        red[wave][0] = p0; red[wave][1] = p1; red[wave][2] = p2;
        red[wave][3] = p3; red[wave][4] = ps;
    }
    __syncthreads();
    if (tid == 0) {
        float q0 = 0, q1 = 0, q2 = 0, q3 = 0, q4 = 0;
        #pragma unroll
        for (int w = 0; w < 8; ++w) {
            q0 += red[w][0]; q1 += red[w][1]; q2 += red[w][2];
            q3 += red[w][3]; q4 += red[w][4];
        }
        tb[s * 4 + 0] = q0; tb[s * 4 + 1] = q1;
        tb[s * 4 + 2] = q2; tb[s * 4 + 3] = q3;
        rsb[s] = q4;
    }
}

// --------------------------------------------------------------------------
// Stage one [128 x 32] f32 tile into LDS as hi/lo bf16 (bf16x3 split)
// --------------------------------------------------------------------------
__device__ __forceinline__ void stage_tile(
    const float* __restrict__ src, int ldsrc,
    __bf16* dstH, __bf16* dstL, int tid)
{
    const int r0 = tid >> 3;          // 0..31
    const int kq = (tid & 7) * 4;     // 0,4,...,28
    #pragma unroll
    for (int it = 0; it < 4; ++it) {
        const int row = r0 + it * 32;
        float4 v = *(const float4*)(src + (long)row * ldsrc + kq);
        float f[4] = {v.x, v.y, v.z, v.w};
        v4bf h, l;
        #pragma unroll
        for (int e = 0; e < 4; ++e) {
            __bf16 hh = (__bf16)f[e];
            h[e] = hh;
            l[e] = (__bf16)(f[e] - (float)hh);
        }
        *(v4bf*)(dstH + row * LDK + kq) = h;
        *(v4bf*)(dstL + row * LDK + kq) = l;
    }
}

// --------------------------------------------------------------------------
// Kernel 2: main GEMM  out = x@W^T (bf16x3 WMMA) + fused rank-4/zc/bias epilogue
// --------------------------------------------------------------------------
__global__ __launch_bounds__(256) void zclora_gemm(
    const float* __restrict__ x, const float* __restrict__ W,
    const float* __restrict__ A, const float* __restrict__ tbuf,
    const float* __restrict__ rsbuf, const float* __restrict__ zcw_p,
    const float* __restrict__ zcb_p, const float* __restrict__ bias,
    float* __restrict__ out, int K, int N)
{
    __shared__ __bf16 sA[2][2][BM * LDK];   // [buf][hi/lo]
    __shared__ __bf16 sB[2][2][BN * LDK];

    const int tid  = threadIdx.x;
    const int lane = tid & 31;
    const int wave = tid >> 5;
    const int wM   = wave >> 2;      // 0..1 (64-row wave tile)
    const int wN   = wave & 3;       // 0..3 (32-col wave tile)
    const int l16  = lane & 15;
    const int lh   = lane >> 4;

    const int bM = blockIdx.y * BM;
    const int bN = blockIdx.x * BN;

    v8f acc[4][2];
    const v8f zf = {0.f, 0.f, 0.f, 0.f, 0.f, 0.f, 0.f, 0.f};
    #pragma unroll
    for (int mi = 0; mi < 4; ++mi)
        #pragma unroll
        for (int ni = 0; ni < 2; ++ni)
            acc[mi][ni] = zf;

    const int nk = K / BK;

    stage_tile(x + (long)bM * K, K, &sA[0][0][0], &sA[0][1][0], tid);
    stage_tile(W + (long)bN * K, K, &sB[0][0][0], &sB[0][1][0], tid);
    __syncthreads();

    for (int kt = 0; kt < nk; ++kt) {
        const int buf = kt & 1;
        if (kt + 1 < nk) {
            const int nb = buf ^ 1;
            const int k0 = (kt + 1) * BK;
            stage_tile(x + (long)bM * K + k0, K, &sA[nb][0][0], &sA[nb][1][0], tid);
            stage_tile(W + (long)bN * K + k0, K, &sB[nb][0][0], &sB[nb][1][0], tid);
        }
        if (kt + 2 < nk) {   // global_prefetch_b8 two stages ahead
            const long pofs = (long)(tid >> 3) * K + (kt + 2) * BK + (tid & 7) * 4;
            __builtin_prefetch(W + (long)bN * K + pofs, 0, 1);
            __builtin_prefetch(x + (long)bM * K + pofs, 0, 1);
        }

        // ---- A fragments (16-bit A 16x32 layout: K = {lh*8..+7, 16+lh*8..+7}) ----
        Frag ah[4], al[4], bh[2], bl[2];
        #pragma unroll
        for (int mi = 0; mi < 4; ++mi) {
            const int row  = wM * 64 + mi * 16 + l16;
            const int base = row * LDK + lh * 8;
            const __bf16* pH = &sA[buf][0][base];
            const __bf16* pL = &sA[buf][1][base];
            ah[mi].h[0] = *(const v8bf*)pH;
            ah[mi].h[1] = *(const v8bf*)(pH + 16);
            al[mi].h[0] = *(const v8bf*)pL;
            al[mi].h[1] = *(const v8bf*)(pL + 16);
        }
        // ---- B fragments (16-bit B 32x16 layout: K = {lh*8..+7, 16+lh*8..+7}, col = l16) ----
        #pragma unroll
        for (int ni = 0; ni < 2; ++ni) {
            const int col  = wN * 32 + ni * 16 + l16;
            const int base = col * LDK + lh * 8;
            const __bf16* pH = &sB[buf][0][base];
            const __bf16* pL = &sB[buf][1][base];
            bh[ni].h[0] = *(const v8bf*)pH;
            bh[ni].h[1] = *(const v8bf*)(pH + 16);
            bl[ni].h[0] = *(const v8bf*)pL;
            bl[ni].h[1] = *(const v8bf*)(pL + 16);
        }
        // ---- bf16x3: hi*hi + hi*lo + lo*hi, f32 accumulate ----
        #pragma unroll
        for (int mi = 0; mi < 4; ++mi) {
            #pragma unroll
            for (int ni = 0; ni < 2; ++ni) {
                acc[mi][ni] = __builtin_amdgcn_wmma_f32_16x16x32_bf16(
                    false, ah[mi].v, false, bh[ni].v, (short)0, acc[mi][ni], false, false);
                acc[mi][ni] = __builtin_amdgcn_wmma_f32_16x16x32_bf16(
                    false, ah[mi].v, false, bl[ni].v, (short)0, acc[mi][ni], false, false);
                acc[mi][ni] = __builtin_amdgcn_wmma_f32_16x16x32_bf16(
                    false, al[mi].v, false, bh[ni].v, (short)0, acc[mi][ni], false, false);
            }
        }
        __syncthreads();
    }

    // ---- fused epilogue: + 8*zc_w*(t@A^T) + zc_b*rowsum + bias ----
    const float s_eff = 8.0f * zcw_p[0];
    const float zcb   = zcb_p[0];
    #pragma unroll
    for (int mi = 0; mi < 4; ++mi) {
        #pragma unroll
        for (int ni = 0; ni < 2; ++ni) {
            const int n = bN + wN * 32 + ni * 16 + l16;
            const float4 a4 = *(const float4*)(A + n * 4);
            const float  bn = bias[n];
            #pragma unroll
            for (int r = 0; r < 8; ++r) {
                // C/D layout: VGPR r holds M = r (lanes 0-15) / r+8 (lanes 16-31)
                const int m = bM + wM * 64 + mi * 16 + lh * 8 + r;
                const float4 t4 = *(const float4*)(tbuf + m * 4);
                const float corr =
                    s_eff * (t4.x * a4.x + t4.y * a4.y + t4.z * a4.z + t4.w * a4.w) +
                    zcb * rsbuf[m] + bn;
                out[(long)m * N + n] = acc[mi][ni][r] + corr;
            }
        }
    }
}

// --------------------------------------------------------------------------
extern "C" void kernel_launch(void* const* d_in, const int* in_sizes, int n_in,
                              void* d_out, int out_size, void* d_ws, size_t ws_size,
                              hipStream_t stream) {
    const float* x    = (const float*)d_in[0];
    const float* W    = (const float*)d_in[1];
    const float* A    = (const float*)d_in[2];
    const float* B    = (const float*)d_in[3];
    const float* zc_w = (const float*)d_in[4];
    const float* zc_b = (const float*)d_in[5];
    const float* bias = (const float*)d_in[6];
    float* out = (float*)d_out;

    const int N = in_sizes[6];            // 4096 (d_out)
    const int K = in_sizes[3] / 4;        // 4096 (d_in, B is [4, d_in])
    const int M = in_sizes[0] / K;        // 512  (batch*seq)

    float* tb = (float*)d_ws;             // [M][4]
    float* rs = tb + (size_t)M * 4;       // [M]

    zclora_rank4<<<M, 256, 0, stream>>>(x, B, tb, rs, K);
    zclora_gemm<<<dim3(N / BN, M / BM), 256, 0, stream>>>(
        x, W, A, tb, rs, zc_w, zc_b, bias, out, K, N);
}
